// VGG_Cifar10_9629316678097
// MI455X (gfx1250) — 8-bit matrix operand order repaired and checked by reading, compiled, not run on hardware
//
#include <hip/hip_runtime.h>


typedef __attribute__((ext_vector_type(8))) int v8i;
typedef int v4ig __attribute__((vector_size(16)));

#define EPSBN 1e-5f

#define GAS __attribute__((address_space(1)))
#define LAS __attribute__((address_space(3)))

#if __has_builtin(__builtin_amdgcn_global_load_async_to_lds_b128) && \
    __has_builtin(__builtin_amdgcn_global_load_async_to_lds_b8)
#define HAVE_ASYNC 1
#else
#define HAVE_ASYNC 0
#endif

__device__ __forceinline__ void asy_b128(const void* g, void* l) {
#if HAVE_ASYNC
  __builtin_amdgcn_global_load_async_to_lds_b128(
      (GAS v4ig*)(unsigned long long)g,
      (LAS v4ig*)(unsigned int)(unsigned long long)l, 0, 0);
#else
  *(int4*)l = *(const int4*)g;
#endif
}
__device__ __forceinline__ void asy_b8(const void* g, void* l) {
#if HAVE_ASYNC
  __builtin_amdgcn_global_load_async_to_lds_b8(
      (GAS char*)(unsigned long long)g,
      (LAS char*)(unsigned int)(unsigned long long)l, 0, 0);
#else
  *(signed char*)l = *(const signed char*)g;
#endif
}
__device__ __forceinline__ void asy_wait() {
#if HAVE_ASYNC
#if __has_builtin(__builtin_amdgcn_s_wait_asynccnt)
  __builtin_amdgcn_s_wait_asynccnt(0);
#else
  asm volatile("s_wait_asynccnt 0x0" ::: "memory");
#endif
#endif
}

__global__ __launch_bounds__(256) void zero_f32(float* p, int n) {
  int i = blockIdx.x * 256 + threadIdx.x;
  if (i < n) p[i] = 0.f;
}

__global__ __launch_bounds__(256) void pack_sign(const float* __restrict__ w,
                                                 signed char* __restrict__ wp,
                                                 int K, int Kpad) {
  int co = blockIdx.y;
  int k = blockIdx.x * 256 + threadIdx.x;
  if (k >= Kpad) return;
  signed char v = 0;
  if (k < K) {
    float f = w[(size_t)co * K + k];
    v = (signed char)((f > 0.f) - (f < 0.f));
  }
  wp[(size_t)co * Kpad + k] = v;
}

__device__ __forceinline__ float conv1_val(const float* __restrict__ x,
                                           const float* __restrict__ wsgn,
                                           int b, int y, int xx) {
  float acc = 0.f;
#pragma unroll
  for (int ci = 0; ci < 3; ++ci)
#pragma unroll
    for (int ky = 0; ky < 3; ++ky) {
      int iy = y + ky - 1;
      if ((unsigned)iy >= 32u) continue;
#pragma unroll
      for (int kx = 0; kx < 3; ++kx) {
        int ix = xx + kx - 1;
        if ((unsigned)ix >= 32u) continue;
        acc += wsgn[ci * 9 + ky * 3 + kx] *
               x[(((size_t)b * 3 + ci) << 10) + (iy << 5) + ix];
      }
    }
  return acc;
}

__global__ __launch_bounds__(256) void conv1_stats(const float* __restrict__ x,
                                                   const float* __restrict__ cw1,
                                                   float* __restrict__ stats) {
  __shared__ float wsgn[27];
  __shared__ float rs[256];
  __shared__ float rq[256];
  int c = blockIdx.y;
  int t = threadIdx.x;
  if (t < 27) {
    float f = cw1[c * 27 + t];
    wsgn[t] = (float)((f > 0.f) - (f < 0.f));
  }
  __syncthreads();
  int j = blockIdx.x * 256 + t;
  int b = j >> 10, y = (j >> 5) & 31, xx = j & 31;
  float v = conv1_val(x, wsgn, b, y, xx);
  rs[t] = v;
  rq[t] = v * v;
  __syncthreads();
  for (int s = 128; s > 0; s >>= 1) {
    if (t < s) { rs[t] += rs[t + s]; rq[t] += rq[t + s]; }
    __syncthreads();
  }
  if (t == 0) {
    atomicAdd(&stats[2 * c], rs[0]);
    atomicAdd(&stats[2 * c + 1], rq[0]);
  }
}

__global__ __launch_bounds__(256) void conv1_apply(const float* __restrict__ x,
                                                   const float* __restrict__ cw1,
                                                   const float* __restrict__ stats,
                                                   const float* __restrict__ g,
                                                   const float* __restrict__ bta,
                                                   signed char* __restrict__ act) {
  __shared__ float wsgn[27];
  int c = blockIdx.y;
  int t = threadIdx.x;
  if (t < 27) {
    float f = cw1[c * 27 + t];
    wsgn[t] = (float)((f > 0.f) - (f < 0.f));
  }
  __syncthreads();
  int j = blockIdx.x * 256 + t;
  int b = j >> 10, y = (j >> 5) & 31, xx = j & 31;
  float v = conv1_val(x, wsgn, b, y, xx);
  const float cnt = 131072.f;
  float m = stats[2 * c] / cnt;
  float var = stats[2 * c + 1] / cnt - m * m;
  float xn = (v - m) * rsqrtf(var + EPSBN) * g[c] + bta[c];
  act[(((size_t)b * 384 + c) << 10) + (y << 5) + xx] =
      (signed char)((xn > 0.f) - (xn < 0.f));
}

template <int KS>
__global__ __launch_bounds__(256) void bin_gemm_wmma(
    const signed char* __restrict__ act, const signed char* __restrict__ wp,
    short* __restrict__ out, const signed char* __restrict__ zpage, int Cin,
    int Cout, int Kpad, int logW) {
  __shared__ signed char lA[2][128 * 64];
  __shared__ signed char lB[2][64 * 64];

  const int t = threadIdx.x;
  const int wv = t >> 5;
  const int l = t & 31;
  const int llo = l & 15;
  const int lhi = l >> 4;
  const int n0 = blockIdx.x * 64;
  const int co0 = blockIdx.y * 128;
  const int W = 1 << logW;
  const int HW = 1 << (2 * logW);

  v8i acc[4] = {v8i{}, v8i{}, v8i{}, v8i{}};

  const int ar = t >> 1;
  const int ah = (t & 1) * 32;
  const int bcol = t & 63;
  const int bkk = (t >> 6) * 16;

  const signed char* aSrc = wp + (size_t)(co0 + ar) * Kpad + ah;

  int n = n0 + bcol;
  int b = n >> (2 * logW);
  int rest = n & (HW - 1);
  int y = rest >> logW;
  int xx = rest & (W - 1);
  const signed char* actBase = act + ((size_t)(b * Cin) << (2 * logW));
  int ci = bkk / 9;
  int rm = bkk - ci * 9;
  int ky = rm / 3;
  int kx = rm - ky * 3;

  auto stageTile = [&](int k0, int sbuf) {
    asy_b128(aSrc + k0, &lA[sbuf][ar * 64 + ah]);
    asy_b128(aSrc + k0 + 16, &lA[sbuf][ar * 64 + ah + 16]);
    __builtin_prefetch(aSrc + k0 + 128, 0, 1);
    if constexpr (KS == 1) {
      asy_b128(act + (size_t)n * Cin + k0 + bkk, &lB[sbuf][bcol * 64 + bkk]);
    } else {
      int ciI = ci, kyI = ky, kxI = kx;
      signed char* ldst = &lB[sbuf][bcol * 64 + bkk];
#pragma unroll
      for (int i = 0; i < 16; ++i) {
        int iy = y + kyI - 1;
        int ix = xx + kxI - 1;
        bool ok = ((unsigned)iy < (unsigned)W) & ((unsigned)ix < (unsigned)W);
        const signed char* src =
            ok ? actBase + ((size_t)ciI << (2 * logW)) + (iy << logW) + ix
               : zpage;
        asy_b8(src, ldst + i);
        ++kxI;
        if (kxI == 3) {
          kxI = 0;
          ++kyI;
          if (kyI == 3) { kyI = 0; ++ciI; }
        }
      }
    }
  };
  auto advance64 = [&]() {
    ci += 7;
    ++kx;
    if (kx == 3) {
      kx = 0;
      ++ky;
      if (ky == 3) { ky = 0; ++ci; }
    }
  };

  stageTile(0, 0);
  advance64();
  asy_wait();
  __syncthreads();

  int buf = 0;
  for (int k0 = 0; k0 < Kpad; k0 += 64) {
    if (k0 + 64 < Kpad) {
      stageTile(k0 + 64, buf ^ 1);
      advance64();
    }
    union { v8i v; unsigned long long q[4]; } af;
    {
      const signed char* ap = &lA[buf][(wv * 16 + llo) * 64 + lhi * 8];
      af.q[0] = *(const unsigned long long*)(ap + 0);
      af.q[1] = *(const unsigned long long*)(ap + 16);
      af.q[2] = *(const unsigned long long*)(ap + 32);
      af.q[3] = *(const unsigned long long*)(ap + 48);
    }
    union BF { v8i v; unsigned long long q[4]; } bf[4];
#pragma unroll
    for (int j = 0; j < 4; ++j) {
      const signed char* bp = &lB[buf][(j * 16 + llo) * 64 + lhi * 8];
      bf[j].q[0] = *(const unsigned long long*)(bp + 0);
      bf[j].q[1] = *(const unsigned long long*)(bp + 16);
      bf[j].q[2] = *(const unsigned long long*)(bp + 32);
      bf[j].q[3] = *(const unsigned long long*)(bp + 48);
    }
#pragma unroll
    for (int j = 0; j < 4; ++j)
      acc[j] = __builtin_amdgcn_wmma_i32_16x16x64_iu8(true, af.v, true,
                                                      bf[j].v, acc[j], false,
                                                      false);
    asy_wait();
    __syncthreads();
    buf ^= 1;
  }

#pragma unroll
  for (int j = 0; j < 4; ++j) {
    int nn = n0 + j * 16 + llo;
    int bb = nn >> (2 * logW);
    int rr = nn & (HW - 1);
#pragma unroll
    for (int r = 0; r < 8; ++r) {
      int co = co0 + wv * 16 + lhi * 8 + r;
      out[(((size_t)bb * Cout + co) << (2 * logW)) + rr] = (short)acc[j][r];
    }
  }
}

__global__ __launch_bounds__(256) void maxpool_i16(const short* __restrict__ in,
                                                   short* __restrict__ out,
                                                   int C, int logWo, int B) {
  int c = blockIdx.y;
  int j = blockIdx.x * 256 + threadIdx.x;
  int Wo = 1 << logWo, HWo = 1 << (2 * logWo);
  if (j >= B * HWo) return;
  int b = j >> (2 * logWo);
  int s = j & (HWo - 1);
  int yo = s >> logWo, xo = s & (Wo - 1);
  int logWi = logWo + 1;
  size_t ib = (((size_t)b * C + c) << (2 * logWi)) +
              ((size_t)(2 * yo) << logWi) + 2 * xo;
  int Wi = 1 << logWi;
  short a0 = in[ib], a1 = in[ib + 1], a2 = in[ib + Wi], a3 = in[ib + Wi + 1];
  short m0 = a0 > a1 ? a0 : a1;
  short m1 = a2 > a3 ? a2 : a3;
  out[(((size_t)b * C + c) << (2 * logWo)) + s] = m0 > m1 ? m0 : m1;
}

__global__ __launch_bounds__(256) void bn_stats_i16(const short* __restrict__ v,
                                                    float* __restrict__ stats,
                                                    int C, int count,
                                                    int logHW) {
  __shared__ float rs[256];
  __shared__ float rq[256];
  int c = blockIdx.x;
  int t = threadIdx.x;
  int HWm = (1 << logHW) - 1;
  float s = 0.f, q = 0.f;
  for (int i = t; i < count; i += 256) {
    int b = i >> logHW;
    int sp = i & HWm;
    float f = (float)v[(((size_t)b * C + c) << logHW) + sp];
    s += f;
    q += f * f;
  }
  rs[t] = s;
  rq[t] = q;
  __syncthreads();
  for (int st = 128; st > 0; st >>= 1) {
    if (t < st) { rs[t] += rs[t + st]; rq[t] += rq[t + st]; }
    __syncthreads();
  }
  if (t == 0) {
    stats[2 * c] = rs[0];
    stats[2 * c + 1] = rq[0];
  }
}

__global__ __launch_bounds__(256) void bn_apply_sign(
    const short* __restrict__ v, const float* __restrict__ stats,
    const float* __restrict__ g, const float* __restrict__ bta,
    signed char* __restrict__ out, int C, int logHW, int B) {
  int c = blockIdx.y;
  int j = blockIdx.x * 256 + threadIdx.x;
  int HW = 1 << logHW;
  if (j >= B * HW) return;
  int b = j >> logHW;
  int s = j & (HW - 1);
  size_t idx = (((size_t)b * C + c) << logHW) + s;
  float cnt = (float)(B << logHW);
  float m = stats[2 * c] / cnt;
  float var = stats[2 * c + 1] / cnt - m * m;
  float xn = ((float)v[idx] - m) * rsqrtf(var + EPSBN) * g[c] + bta[c];
  out[idx] = (signed char)((xn > 0.f) - (xn < 0.f));
}

__global__ __launch_bounds__(256) void fc3_small(const signed char* __restrict__ a,
                                                 const float* __restrict__ fw3,
                                                 float* __restrict__ o) {
  int t = blockIdx.x * 256 + threadIdx.x;
  if (t >= 1280) return;
  int b = t / 10;
  int oo = t - b * 10;
  int s = 0;
  for (int k = 0; k < 1024; ++k) {
    float f = fw3[oo * 1024 + k];
    int sg = (f > 0.f) - (f < 0.f);
    s += sg * (int)a[b * 1024 + k];
  }
  o[t] = (float)s;
}

__global__ __launch_bounds__(128) void final_bn_lsm(const float* __restrict__ f,
                                                    float* __restrict__ out) {
  __shared__ float sm[10];
  __shared__ float sv[10];
  int b = threadIdx.x;
  if (b < 10) {
    float s = 0.f, q = 0.f;
    for (int r = 0; r < 128; ++r) {
      float x = f[r * 10 + b];
      s += x;
      q += x * x;
    }
    float m = s / 128.f;
    sm[b] = m;
    sv[b] = rsqrtf(q / 128.f - m * m + EPSBN);
  }
  __syncthreads();
  float xn[10];
  float mx = -3.4e38f;
  for (int o = 0; o < 10; ++o) {
    xn[o] = (f[b * 10 + o] - sm[o]) * sv[o];
    mx = fmaxf(mx, xn[o]);
  }
  float se = 0.f;
  for (int o = 0; o < 10; ++o) se += expf(xn[o] - mx);
  float lse = mx + logf(se);
  for (int o = 0; o < 10; ++o) out[b * 10 + o] = xn[o] - lse;
}

extern "C" void kernel_launch(void* const* d_in, const int* in_sizes, int n_in,
                              void* d_out, int out_size, void* d_ws,
                              size_t ws_size, hipStream_t stream) {
  (void)in_sizes; (void)n_in; (void)out_size; (void)ws_size;
  const float* x   = (const float*)d_in[0];
  const float* cw1 = (const float*)d_in[1];
  const float* g1  = (const float*)d_in[3];
  const float* b1  = (const float*)d_in[4];
  const float* cw2 = (const float*)d_in[5];
  const float* g2  = (const float*)d_in[7];
  const float* b2  = (const float*)d_in[8];
  const float* cw3 = (const float*)d_in[9];
  const float* g3  = (const float*)d_in[11];
  const float* b3  = (const float*)d_in[12];
  const float* cw4 = (const float*)d_in[13];
  const float* g4  = (const float*)d_in[15];
  const float* b4  = (const float*)d_in[16];
  const float* cw5 = (const float*)d_in[17];
  const float* g5  = (const float*)d_in[19];
  const float* b5  = (const float*)d_in[20];
  const float* cw6 = (const float*)d_in[21];
  const float* g6  = (const float*)d_in[23];
  const float* b6  = (const float*)d_in[24];
  const float* fw1 = (const float*)d_in[25];
  const float* lg1 = (const float*)d_in[27];
  const float* lb1 = (const float*)d_in[28];
  const float* fw2 = (const float*)d_in[29];
  const float* lg2 = (const float*)d_in[31];
  const float* lb2 = (const float*)d_in[32];
  const float* fw3 = (const float*)d_in[33];

  char* ws = (char*)d_ws;
  size_t off = 0;
  auto alloc = [&](size_t bytes) {
    size_t o = off;
    off += (bytes + 255) & ~(size_t)255;
    return o;
  };
  signed char* wp2 = (signed char*)(ws + alloc(384ULL * 3456));
  signed char* wp3 = (signed char*)(ws + alloc(768ULL * 3456));
  signed char* wp4 = (signed char*)(ws + alloc(768ULL * 6912));
  signed char* wp5 = (signed char*)(ws + alloc(1536ULL * 6912));
  signed char* wp6 = (signed char*)(ws + alloc(512ULL * 13824));
  signed char* wf1 = (signed char*)(ws + alloc(1024ULL * 8192));
  signed char* wf2 = (signed char*)(ws + alloc(1024ULL * 1024));
  short* conv16 = (short*)(ws + alloc(128ULL * 384 * 1024 * 2));
  short* pool16 = (short*)(ws + alloc(128ULL * 384 * 256 * 2));
  signed char* actA = (signed char*)(ws + alloc(128ULL * 384 * 1024));
  signed char* actB = (signed char*)(ws + alloc(128ULL * 768 * 256));
  float* stats = (float*)(ws + alloc(2ULL * 1536 * 4));
  float* f3o   = (float*)(ws + alloc(1280ULL * 4));
  signed char* zpage = (signed char*)(ws + alloc(256));

  pack_sign<<<dim3(14, 384), 256, 0, stream>>>(cw2, wp2, 3456, 3456);
  pack_sign<<<dim3(14, 768), 256, 0, stream>>>(cw3, wp3, 3456, 3456);
  pack_sign<<<dim3(27, 768), 256, 0, stream>>>(cw4, wp4, 6912, 6912);
  pack_sign<<<dim3(27, 1536), 256, 0, stream>>>(cw5, wp5, 6912, 6912);
  pack_sign<<<dim3(54, 512), 256, 0, stream>>>(cw6, wp6, 13824, 13824);
  pack_sign<<<dim3(32, 1024), 256, 0, stream>>>(fw1, wf1, 8192, 8192);
  pack_sign<<<dim3(4, 1024), 256, 0, stream>>>(fw2, wf2, 1024, 1024);
  zero_f32<<<1, 256, 0, stream>>>((float*)zpage, 64);
  zero_f32<<<3, 256, 0, stream>>>(stats, 768);

  conv1_stats<<<dim3(512, 384), 256, 0, stream>>>(x, cw1, stats);
  conv1_apply<<<dim3(512, 384), 256, 0, stream>>>(x, cw1, stats, g1, b1, actA);

  bin_gemm_wmma<9><<<dim3(2048, 3), 256, 0, stream>>>(actA, wp2, conv16, zpage, 384, 384, 3456, 5);
  maxpool_i16<<<dim3(128, 384), 256, 0, stream>>>(conv16, pool16, 384, 4, 128);
  bn_stats_i16<<<384, 256, 0, stream>>>(pool16, stats, 384, 128 * 256, 8);
  bn_apply_sign<<<dim3(128, 384), 256, 0, stream>>>(pool16, stats, g2, b2, actB, 384, 8, 128);

  bin_gemm_wmma<9><<<dim3(512, 6), 256, 0, stream>>>(actB, wp3, conv16, zpage, 384, 768, 3456, 4);
  bn_stats_i16<<<768, 256, 0, stream>>>(conv16, stats, 768, 128 * 256, 8);
  bn_apply_sign<<<dim3(128, 768), 256, 0, stream>>>(conv16, stats, g3, b3, actA, 768, 8, 128);

  bin_gemm_wmma<9><<<dim3(512, 6), 256, 0, stream>>>(actA, wp4, conv16, zpage, 768, 768, 6912, 4);
  maxpool_i16<<<dim3(32, 768), 256, 0, stream>>>(conv16, pool16, 768, 3, 128);
  bn_stats_i16<<<768, 256, 0, stream>>>(pool16, stats, 768, 128 * 64, 6);
  bn_apply_sign<<<dim3(32, 768), 256, 0, stream>>>(pool16, stats, g4, b4, actB, 768, 6, 128);

  bin_gemm_wmma<9><<<dim3(128, 12), 256, 0, stream>>>(actB, wp5, conv16, zpage, 768, 1536, 6912, 3);
  bn_stats_i16<<<1536, 256, 0, stream>>>(conv16, stats, 1536, 128 * 64, 6);
  bn_apply_sign<<<dim3(32, 1536), 256, 0, stream>>>(conv16, stats, g5, b5, actA, 1536, 6, 128);

  bin_gemm_wmma<9><<<dim3(128, 4), 256, 0, stream>>>(actA, wp6, conv16, zpage, 1536, 512, 13824, 3);
  maxpool_i16<<<dim3(8, 512), 256, 0, stream>>>(conv16, pool16, 512, 2, 128);
  bn_stats_i16<<<512, 256, 0, stream>>>(pool16, stats, 512, 128 * 16, 4);
  bn_apply_sign<<<dim3(8, 512), 256, 0, stream>>>(pool16, stats, g6, b6, actB, 512, 4, 128);

  bin_gemm_wmma<1><<<dim3(2, 8), 256, 0, stream>>>(actB, wf1, conv16, zpage, 8192, 1024, 8192, 0);
  bn_stats_i16<<<1024, 256, 0, stream>>>(conv16, stats, 1024, 128, 0);
  bn_apply_sign<<<dim3(1, 1024), 256, 0, stream>>>(conv16, stats, lg1, lb1, actA, 1024, 0, 128);

  bin_gemm_wmma<1><<<dim3(2, 8), 256, 0, stream>>>(actA, wf2, conv16, zpage, 1024, 1024, 1024, 0);
  bn_stats_i16<<<1024, 256, 0, stream>>>(conv16, stats, 1024, 128, 0);
  bn_apply_sign<<<dim3(1, 1024), 256, 0, stream>>>(conv16, stats, lg2, lb2, actB, 1024, 0, 128);

  fc3_small<<<5, 256, 0, stream>>>(actB, fw3, f3o);
  final_bn_lsm<<<1, 128, 0, stream>>>(f3o, (float*)d_out);
}
